// MultiHeadSelfAttention_1434519077370
// MI455X (gfx1250) — matrix operand order repaired and checked by reading, compiled, not run on hardware
//
#include <hip/hip_runtime.h>

typedef __attribute__((ext_vector_type(16))) __bf16       v16bf;
typedef __attribute__((ext_vector_type(8)))  float        v8f;
typedef __attribute__((ext_vector_type(4)))  unsigned int u32x4;
typedef __attribute__((ext_vector_type(8)))  int          i32x8;
typedef __attribute__((ext_vector_type(4)))  int          i32x4;
typedef __attribute__((ext_vector_type(4)))  float        f32x4;
typedef unsigned short u16;

#define B_  2
#define T_  2048
#define C_  1024
#define H_  16
#define D_  64
#define M_  (B_*T_)   // 4096 rows of x

#if defined(__has_builtin)
#  if __has_builtin(__builtin_amdgcn_tensor_load_to_lds)
#    define HAVE_TDM 1
#  else
#    define HAVE_TDM 0
#  endif
#else
#  define HAVE_TDM 0
#endif

union Frag   { v16bf v; u32x4 q[2]; };          // 32B WMMA operand fragment
union Pack16 { u32x4 q[2]; u16 s[16]; };        // 16 bf16 repack buffer

__device__ __forceinline__ u16 f2bf(float f) {  // round-to-nearest-even f32 -> bf16
  union { float f; unsigned int u; } c; c.f = f;
  unsigned int u = c.u;
  u += 0x7FFFu + ((u >> 16) & 1u);
  return (u16)(u >> 16);
}

__device__ __forceinline__ v8f wmma_bf16(const Frag& a, const Frag& b, v8f c) {
  // (neg_a, A, neg_b, B, c_mod, C, reuse_a, reuse_b)
  return __builtin_amdgcn_wmma_f32_16x16x32_bf16(false, a.v, false, b.v,
                                                 (short)0, c, false, false);
}

#if HAVE_TDM
// -----------------------------------------------------------------------
// TDM: DMA a 2D tile (tile_d1 rows x tile_d0 elems, 2-byte elements) from
// global [rows x stride0] into LDS at lds_addr, inserting 16B of LDS pad
// after every 128B row (-> stride-72-u16 rows, matches Ks layout).
// D# packing per CDNA5 ISA.
// -----------------------------------------------------------------------
__device__ __forceinline__ void tdm_load_tile_2d(unsigned lds_addr, const void* gptr,
                                                 unsigned tensor_d0, unsigned tensor_d1,
                                                 unsigned tile_d0, unsigned tile_d1,
                                                 unsigned stride0)
{
  unsigned long long ga = (unsigned long long)(size_t)gptr;
  u32x4 g0;
  g0.x = 1u;                                              // count=1 (valid), no gather
  g0.y = lds_addr;                                        // bits 63:32  lds_addr
  g0.z = (unsigned)ga;                                    // bits 95:64  global_addr lo
  g0.w = (unsigned)((ga >> 32) & 0x1FFFFFFu) | (2u << 30);// addr[56:32] | type=2
  i32x8 g1;
  // d0: wg_mask=0 | data_size=1(2B)<<16 | pad_enable<<20 | pad_interval=4(32DW)<<22
  //     | pad_amount=3(4DW=16B)<<25
  g1[0] = (int)((1u << 16) | (1u << 20) | (4u << 22) | (3u << 25));
  g1[1] = (int)((tensor_d0 & 0xFFFFu) << 16);             // abar=0 | tensor_dim0[15:0]
  g1[2] = (int)(((tensor_d0 >> 16) & 0xFFFFu) | ((tensor_d1 & 0xFFFFu) << 16));
  g1[3] = (int)(((tensor_d1 >> 16) & 0xFFFFu) | ((tile_d0 & 0xFFFFu) << 16));
  g1[4] = (int)(tile_d1 & 0xFFFFu);                       // tile_dim1 | tile_dim2=0
  g1[5] = (int)stride0;                                   // tensor_dim0_stride[31:0]
  g1[6] = 0;                                              // stride0 hi | dim1_stride lo
  g1[7] = 0;
  i32x4 z4 = {0, 0, 0, 0};
#if __clang_major__ >= 23
  i32x8 z8 = {};
  __builtin_amdgcn_tensor_load_to_lds(g0, g1, z4, z4, z8, 0);
#else
  __builtin_amdgcn_tensor_load_to_lds(g0, g1, z4, z4, 0);
#endif
}
#endif

// ---------------------------------------------------------------------------
// Tiled GEMM: Y[M,N] = A[M,K] @ W[N,K]^T  (K = C_ = 1024, N = C_)
// Block tile 128x128, 8 waves in 2(M)x4(N), each wave 4x2 of 16x16 WMMA tiles.
// ---------------------------------------------------------------------------
template<bool A_BF16, bool QKV_OUT>
__global__ __launch_bounds__(256)
void gemm_wmma_k(const void* __restrict__ Aptr, const float* __restrict__ W,
                 void* __restrict__ outp, float scale)
{
  __shared__ u16 As[128 * 40];   // 128 rows x 32 k, stride 40 (bank-conflict pad)
  __shared__ u16 Bs[128 * 40];   // 128 out-channels x 32 k

  const int tid  = threadIdx.x;
  const int m0   = blockIdx.x * 128;
  const int n0   = blockIdx.y * 128;
  const int lane = tid & 31, wave = tid >> 5;
  const int wm   = wave >> 2, wn = wave & 3;       // 2 x 4 wave grid
  const int lrow = lane & 15, hi = lane >> 4;
  const int klo  = hi * 8;                          // A-frag K base (ISA layout)
  const int klo2 = hi * 16;                         // unused: B-frag K base is klo, as for A

  v8f acc[4][2] = {};

  const int lr = tid >> 1;            // 0..127: LDS row this thread loads
  const int lc = (tid & 1) * 16;      // 0 / 16:  16-element k segment

  for (int k0 = 0; k0 < C_; k0 += 32) {
    __syncthreads();
    // ---- A tile -> As (convert fp32 -> bf16 if needed) ----
    if (A_BF16) {
      const u32x4* src = (const u32x4*)((const u16*)Aptr + (size_t)(m0 + lr) * C_ + k0 + lc);
      u32x4* dst = (u32x4*)&As[lr * 40 + lc];
      dst[0] = src[0]; dst[1] = src[1];
    } else {
      const f32x4* src = (const f32x4*)((const float*)Aptr + (size_t)(m0 + lr) * C_ + k0 + lc);
      Pack16 p;
      #pragma unroll
      for (int j = 0; j < 4; ++j) {
        f32x4 f = src[j];
        p.s[4*j+0] = f2bf(f.x); p.s[4*j+1] = f2bf(f.y);
        p.s[4*j+2] = f2bf(f.z); p.s[4*j+3] = f2bf(f.w);
      }
      u32x4* dst = (u32x4*)&As[lr * 40 + lc];
      dst[0] = p.q[0]; dst[1] = p.q[1];
    }
    // ---- W tile -> Bs (rows = output channels n; y = x @ W^T) ----
    {
      const f32x4* src = (const f32x4*)(W + (size_t)(n0 + lr) * C_ + k0 + lc);
      Pack16 p;
      #pragma unroll
      for (int j = 0; j < 4; ++j) {
        f32x4 f = src[j];
        p.s[4*j+0] = f2bf(f.x); p.s[4*j+1] = f2bf(f.y);
        p.s[4*j+2] = f2bf(f.z); p.s[4*j+3] = f2bf(f.w);
      }
      u32x4* dst = (u32x4*)&Bs[lr * 40 + lc];
      dst[0] = p.q[0]; dst[1] = p.q[1];
    }
    __syncthreads();

    // ---- fragments + 8 WMMAs per wave per k-step ----
    Frag a[4], b[2];
    #pragma unroll
    for (int mi = 0; mi < 4; ++mi) {
      const u16* base = &As[(wm*64 + mi*16 + lrow) * 40];
      a[mi].q[0] = *(const u32x4*)(base + klo);          // K = klo..klo+7
      a[mi].q[1] = *(const u32x4*)(base + 16 + klo);     // K = 16+klo..23+klo
    }
    #pragma unroll
    for (int ni = 0; ni < 2; ++ni) {
      const u16* base = &Bs[(wn*32 + ni*16 + lrow) * 40];
      b[ni].q[0] = *(const u32x4*)(base + klo);          // K = klo..klo+7
      b[ni].q[1] = *(const u32x4*)(base + 16 + klo);
    }
    #pragma unroll
    for (int mi = 0; mi < 4; ++mi)
      #pragma unroll
      for (int ni = 0; ni < 2; ++ni)
        acc[mi][ni] = wmma_bf16(a[mi], b[ni], acc[mi][ni]);
  }

  // ---- epilogue: C-layout (VGPR r -> M=r+8*hi, N=lane&15) ----
  #pragma unroll
  for (int mi = 0; mi < 4; ++mi) {
    #pragma unroll
    for (int ni = 0; ni < 2; ++ni) {
      #pragma unroll
      for (int r = 0; r < 8; ++r) {
        const int gm = m0 + wm*64 + mi*16 + hi*8 + r;
        const int gn = n0 + wn*32 + ni*16 + lrow;
        const float v = acc[mi][ni][r] * scale;
        if (QKV_OUT) {
          const int b = gm >> 11, t = gm & (T_ - 1);   // T_ = 2048
          const int h = gn >> 6,  d = gn & 63;
          ((u16*)outp)[(((size_t)(b*H_ + h)) * T_ + t) * D_ + d] = f2bf(v);
        } else {
          ((float*)outp)[(size_t)gm * C_ + gn] = v;
        }
      }
    }
  }
}

// ---------------------------------------------------------------------------
// Flash attention (causal, online softmax). Grid: (B*H) * (T/128) blocks.
// 8 waves; wave w owns Q rows [qt*128 + w*16, +16), full D=64.
// K tiles: double-buffered TENSOR_LOAD_TO_LDS (TDM) with s_wait_tensorcnt,
// pad feature reproduces the stride-72 bank-padded layout. V tiles: manual
// transposed copy [d][s]. QK^T and PV on v_wmma_f32_16x16x32_bf16.
// ---------------------------------------------------------------------------
__global__ __launch_bounds__(256)
void flash_attn_k(const u16* __restrict__ Q, const u16* __restrict__ K,
                  const u16* __restrict__ V, u16* __restrict__ O)
{
  __shared__ u16 Ks[2][64 * 72];     // [s][d], stride 72 (TDM pad), double buffer
  __shared__ u16 Vs[64 * 72];        // [d][s] transposed
  __shared__ u16 Ps[8][16 * 72];     // per-wave P tile (C-layout -> A-layout relay)

  const int tid  = threadIdx.x;
  const int bh   = blockIdx.x >> 4;  // T_/128 = 16 q-tiles per (b,h)
  const int qt   = blockIdx.x & 15;
  const int lane = tid & 31, wave = tid >> 5;
  const int lrow = lane & 15, hi = lane >> 4;
  const int klo  = hi * 8, klo2 = hi * 16;

  const size_t bhOff = (size_t)bh * T_ * D_;
  const u16* Qb = Q + bhOff;
  const u16* Kb = K + bhOff;
  const u16* Vb = V + bhOff;

  const int qRow0 = qt * 128 + wave * 16;

  // Q fragments in A-layout straight from global (row-major [T][D] bf16)
  Frag qf[2];
  #pragma unroll
  for (int kc = 0; kc < 2; ++kc) {
    const u16* base = Qb + (size_t)(qRow0 + lrow) * D_ + kc * 32;
    qf[kc].q[0] = *(const u32x4*)(base + klo);
    qf[kc].q[1] = *(const u32x4*)(base + 16 + klo);
  }

  v8f o[4] = {};
  float m_r[8], l_r[8];
  #pragma unroll
  for (int r = 0; r < 8; ++r) { m_r[r] = -1e30f; l_r[r] = 0.f; }

  const int ls   = tid >> 2;          // 0..63: key row this thread stages
  const int lc2  = (tid & 3) * 16;    // 0/16/32/48: d segment
  const int sEnd = qt * 128 + 128;    // causal block bound

#if HAVE_TDM
  // Prologue: post the DMA for the first K tile.
  if (wave == 0)
    tdm_load_tile_2d((unsigned)(size_t)(void*)&Ks[0][0], Kb,
                     D_, T_, D_, 64, D_);
#endif

  for (int s0 = 0; s0 < sEnd; s0 += 64) {
    const int buf = (s0 >> 6) & 1;
    __syncthreads();                  // previous tile fully consumed
    { // V tile -> LDS transposed [d][s] (TDM cannot transpose)
      Pack16 p;
      const u32x4* src = (const u32x4*)(Vb + (size_t)(s0 + ls) * D_ + lc2);
      p.q[0] = src[0]; p.q[1] = src[1];
      #pragma unroll
      for (int j = 0; j < 16; ++j) Vs[(lc2 + j) * 72 + ls] = p.s[j];
    }
#if HAVE_TDM
    if (wave == 0) {
      if (s0 + 64 < sEnd) {           // post DMA for next tile into other buffer,
        tdm_load_tile_2d((unsigned)(size_t)(void*)&Ks[buf ^ 1][0],
                         Kb + (size_t)(s0 + 64) * D_, D_, T_, D_, 64, D_);
        __builtin_amdgcn_s_wait_tensorcnt((short)1);  // current tile landed
      } else {
        __builtin_amdgcn_s_wait_tensorcnt((short)0);
      }
    }
#else
    { // fallback: cooperative K tile copy
      const u32x4* src = (const u32x4*)(Kb + (size_t)(s0 + ls) * D_ + lc2);
      u32x4* dst = (u32x4*)&Ks[buf][ls * 72 + lc2];
      dst[0] = src[0]; dst[1] = src[1];
    }
#endif
    if (s0 + 64 < sEnd)               // prefetch next V tile (global_prefetch_b8)
      __builtin_prefetch(Vb + (size_t)(s0 + 64 + ls) * D_ + lc2, 0, 1);
    __syncthreads();                  // K (TDM) + V (ds) visible to all waves

    const bool active = (s0 <= qRow0);   // tiles fully above diagonal: skip
    if (active) {
      // ---- S = Q @ K^T : 4 n-tiles x 2 k-chunks = 8 WMMAs ----
      v8f s_acc[4] = {};
      #pragma unroll
      for (int ni = 0; ni < 4; ++ni) {
        #pragma unroll
        for (int kc = 0; kc < 2; ++kc) {
          Frag bk;
          const u16* base = &Ks[buf][(ni*16 + lrow) * 72 + kc * 32];
          bk.q[0] = *(const u32x4*)(base + klo);
          bk.q[1] = *(const u32x4*)(base + 16 + klo);
          s_acc[ni] = wmma_bf16(qf[kc], bk, s_acc[ni]);
        }
      }
      // ---- causal mask (boundary tiles only) ----
      if (qRow0 - s0 < 64) {
        #pragma unroll
        for (int ni = 0; ni < 4; ++ni) {
          const int s = s0 + ni*16 + lrow;
          #pragma unroll
          for (int r = 0; r < 8; ++r) {
            const int t = qRow0 + hi*8 + r;
            if (s > t) s_acc[ni][r] = -1e9f;
          }
        }
      }
      // ---- online softmax (rows split across 16-lane halves) ----
      #pragma unroll
      for (int r = 0; r < 8; ++r) {
        float rmax = s_acc[0][r];
        #pragma unroll
        for (int ni = 1; ni < 4; ++ni) rmax = fmaxf(rmax, s_acc[ni][r]);
        #pragma unroll
        for (int off = 8; off >= 1; off >>= 1)
          rmax = fmaxf(rmax, __shfl_xor(rmax, off, 32));
        const float mnew  = fmaxf(m_r[r], rmax);
        const float alpha = __expf(m_r[r] - mnew);
        float rsum = 0.f;
        #pragma unroll
        for (int ni = 0; ni < 4; ++ni) {
          const float p = __expf(s_acc[ni][r] - mnew);
          s_acc[ni][r] = p;
          rsum += p;
        }
        #pragma unroll
        for (int off = 8; off >= 1; off >>= 1)
          rsum += __shfl_xor(rsum, off, 32);
        l_r[r] = l_r[r] * alpha + rsum;
        m_r[r] = mnew;
        #pragma unroll
        for (int dt = 0; dt < 4; ++dt) o[dt][r] *= alpha;
        #pragma unroll
        for (int ni = 0; ni < 4; ++ni)
          Ps[wave][(hi*8 + r) * 72 + ni*16 + lrow] = f2bf(s_acc[ni][r]);
      }
    }
    __syncthreads();   // make Ps visible (and keep barrier counts uniform)

    if (active) {
      // ---- O += P @ V : 4 d-tiles x 2 k-chunks = 8 WMMAs ----
      #pragma unroll
      for (int kc = 0; kc < 2; ++kc) {
        Frag pf;
        const u16* pb = &Ps[wave][lrow * 72 + kc * 32];
        pf.q[0] = *(const u32x4*)(pb + klo);
        pf.q[1] = *(const u32x4*)(pb + 16 + klo);
        #pragma unroll
        for (int dt = 0; dt < 4; ++dt) {
          Frag vf;
          const u16* vb = &Vs[(dt*16 + lrow) * 72 + kc * 32];
          vf.q[0] = *(const u32x4*)(vb + klo);
          vf.q[1] = *(const u32x4*)(vb + 16 + klo);
          o[dt] = wmma_bf16(pf, vf, o[dt]);
        }
      }
    }
  }

  // ---- epilogue: O/l -> bf16 [B,T,C] ----
  const int b = bh / H_, h = bh % H_;
  #pragma unroll
  for (int dt = 0; dt < 4; ++dt) {
    #pragma unroll
    for (int r = 0; r < 8; ++r) {
      const int t = qRow0 + hi*8 + r;
      const float v = o[dt][r] / l_r[r];
      O[((size_t)b * T_ + t) * C_ + h * D_ + dt*16 + lrow] = f2bf(v);
    }
  }
}

// ---------------------------------------------------------------------------
extern "C" void kernel_launch(void* const* d_in, const int* in_sizes, int n_in,
                              void* d_out, int out_size, void* d_ws, size_t ws_size,
                              hipStream_t stream)
{
  (void)in_sizes; (void)n_in; (void)out_size; (void)ws_size;
  const float* x  = (const float*)d_in[0];
  const float* wq = (const float*)d_in[1];
  const float* wk = (const float*)d_in[2];
  const float* wv = (const float*)d_in[3];
  const float* wo = (const float*)d_in[4];

  char* ws = (char*)d_ws;
  const size_t qkvBytes = (size_t)B_ * H_ * T_ * D_ * sizeof(u16);  // 8 MB each
  u16* Qw = (u16*)(ws);
  u16* Kw = (u16*)(ws + 1 * qkvBytes);
  u16* Vw = (u16*)(ws + 2 * qkvBytes);
  u16* Ow = (u16*)(ws + 3 * qkvBytes);   // total 32 MB workspace

  dim3 grid(M_ / 128, C_ / 128), block(256);
  // Q gets the 1/sqrt(D) = 1/8 scale folded in.
  gemm_wmma_k<false, true><<<grid, block, 0, stream>>>(x, wq, Qw, 0.125f);
  gemm_wmma_k<false, true><<<grid, block, 0, stream>>>(x, wk, Kw, 1.0f);
  gemm_wmma_k<false, true><<<grid, block, 0, stream>>>(x, wv, Vw, 1.0f);

  flash_attn_k<<<dim3(B_ * H_ * (T_ / 128)), block, 0, stream>>>(Qw, Kw, Vw, Ow);

  gemm_wmma_k<true, false><<<grid, block, 0, stream>>>(Ow, wo, d_out, 1.0f);
}
